// MotionGatedDualPathAdapter_29283087024828
// MI455X (gfx1250) — matrix operand order repaired and checked by reading, not compiled, not run on hardware
//
#include <hip/hip_runtime.h>
#include <hip/hip_bf16.h>

// ---------------------------------------------------------------- constants
#define BB      32
#define TT      8
#define NPATCH  196
#define CDIM    768
#define ADIM    192
#define HWD     14
#define MROWS   (BB * TT * NPATCH)      // 50176
#define MA      ((size_t)MROWS * ADIM)  // 9,633,792 elems
#define NBLK    192                     // N column-panel width

typedef __attribute__((ext_vector_type(16))) __bf16 v16bf;
typedef __attribute__((ext_vector_type(8)))  __bf16 v8bf;
typedef __attribute__((ext_vector_type(8)))  float  v8f;

__device__ __forceinline__ float gelu_f(float x) {
    return 0.5f * x * (1.0f + erff(x * 0.70710678118654752f));
}

// Row offset: MODE 0 = plain row-major, MODE 1 = patch rows inside x/out
template<int MODE>
__device__ __forceinline__ size_t row_off(int m, int ld) {
    if (MODE == 0) return (size_t)m * (size_t)ld;
    int bt = m / NPATCH;
    int n  = m - bt * NPATCH;
    return ((size_t)bt * 197 + (size_t)n + 1) * (size_t)ld;
}

// --------------------------------------------------- weight pre-pack kernel
// Pack B[KP x N] into WMMA-fragment order, bf16:
//   P[(((tn*KB + kb)*32 + lane)*16 + i)] = B[k][n],
//   n = tn*16 + (lane&15), k = kb*32 + (lane>>4)*16 + i, KB = KP/32.
// A wave's 16x16x32 B-fragment is then 32 contiguous lanes x 32 bytes.
// BTRANS: 0 -> B[k][n] = W[k*ldb+n], 1 -> B[k][n] = W[n*ldb+k].
template<int BTRANS>
__global__ __launch_bounds__(256)
void pack_b_k(const float* __restrict__ W, int ldb, __bf16* __restrict__ P,
              int KP, int total /* = N*KP */)
{
    const int o = blockIdx.x * 256 + threadIdx.x;
    if (o >= total) return;
    const int KB   = KP >> 5;
    const int i    = o & 15;
    const int l    = (o >> 4) & 31;
    const int rest = o >> 9;
    const int kb   = rest % KB;
    const int tn   = rest / KB;
    const int n = tn * 16 + (l & 15);
    const int k = kb * 32 + (l >> 4) * 16 + i;
    const float v = BTRANS ? W[(size_t)n * ldb + k] : W[(size_t)k * ldb + n];
    P[o] = (__bf16)v;
}

// ------------------------------------------------------------- GEMM kernel
// Row-panel WMMA GEMM: Out[M,N] = A[M,KP] * B[KP,N] (+bias) (+resid).
// Block: 256 threads = 8 waves; owns 32 rows of A staged ONCE into LDS as
// bf16 (one barrier per block). B fragments are loaded straight from the
// pre-packed global bf16 buffer (L2-resident) -- no B staging, no K-loop
// barriers. Waves: wm = wave&1 (2x16 rows), wn = wave>>1 (4x48 cols).
// Inner step: 2x ds_load_b128 (A frag) + 6x global_load_b128 (B) + 3x WMMA.
//
// AMODE: 0 plain rows, 1 x-patch rows, 2 temporal-diff of A (ph[t]-ph[t-1])
// RMODE: 0 none, 1 plain resid, 2 x-patch resid, 3 diff-of-resid
// OMODE: output row mapping (0 plain, 1 x-patch)
template<int KP, int AMODE, int BIAS, int RMODE, int OMODE>
__global__ __launch_bounds__(256)
void gemm_panel_wmma(const float* __restrict__ A, int lda,
                     const __bf16* __restrict__ Bp,
                     const float* __restrict__ bias,
                     const float* __restrict__ resid, int ldr,
                     float* __restrict__ Out, int ldo, int N)
{
    __shared__ __bf16 lsA[32 * KP];        // KP=768 -> 48 KB, KP=192 -> 12 KB
    constexpr int KB = KP / 32;

    const int tid  = threadIdx.x;
    const int lane = tid & 31;
    const int wave = tid >> 5;
    const int wm   = wave & 1;             // 2 waves in M
    const int wn   = wave >> 1;            // 4 waves in N
    const int m0   = blockIdx.x * 32;

    // ---- stage the 32 x KP A panel once (f32 -> bf16), float4 loads
    {
        const int f4n = 32 * KP / 4;
        for (int j = tid; j < f4n; j += 256) {
            const int row = j / (KP / 4);
            const int c4  = (j - row * (KP / 4)) * 4;
            const int m   = m0 + row;
            float4 v;
            if constexpr (AMODE == 2) {
                const int t = (m / NPATCH) % TT;
                if (t == 0) {
                    v = make_float4(0.f, 0.f, 0.f, 0.f);
                } else {
                    const float4 a = *(const float4*)(A + (size_t)m * lda + c4);
                    const float4 b = *(const float4*)(A + (size_t)(m - NPATCH) * lda + c4);
                    v = make_float4(a.x - b.x, a.y - b.y, a.z - b.z, a.w - b.w);
                }
            } else {
                v = *(const float4*)(A + row_off<AMODE>(m, lda) + c4);
            }
            __bf16* d = lsA + row * KP + c4;
            d[0] = (__bf16)v.x; d[1] = (__bf16)v.y;
            d[2] = (__bf16)v.z; d[3] = (__bf16)v.w;
        }
    }
    __syncthreads();                       // the ONLY barrier in this kernel

    const int half = lane >> 4;
    const __bf16* arow = lsA + (wm * 16 + (lane & 15)) * KP;

    for (int n0 = 0; n0 < N; n0 += NBLK) {
        v8f acc[3] = {};
        const int tn0 = (n0 >> 4) + wn * 3;

        for (int kb = 0; kb < KB; ++kb) {
            // A fragment (contiguous k = 16*half + i, matching packed B) from LDS
            v16bf af;
            {
                const __bf16* p = arow + kb * 32;
                const v8bf lo = *(const v8bf*)(p + 16 * half);
                const v8bf hi = *(const v8bf*)(p + 16 * half + 8);
#pragma unroll
                for (int i = 0; i < 8; ++i) { af[i] = lo[i]; af[i + 8] = hi[i]; }
            }
            // 3 B fragments straight from packed global (32B/lane, coalesced)
#pragma unroll
            for (int s = 0; s < 3; ++s) {
                const size_t off = (((size_t)(tn0 + s) * KB + kb) * 32 + lane) * 16;
                const v16bf bfr = *(const v16bf*)(Bp + off);
                acc[s] = __builtin_amdgcn_wmma_f32_16x16x32_bf16(
                             false, af, false, bfr, (short)0, acc[s], false, false);
            }
        }

        // ---- epilogue: C elem e -> M = e + 8*half, N = lane&15
        const int cn = lane & 15;
#pragma unroll
        for (int s = 0; s < 3; ++s) {
            const int col = n0 + wn * 48 + s * 16 + cn;
#pragma unroll
            for (int e = 0; e < 8; ++e) {
                const int mrow = m0 + wm * 16 + e + 8 * half;
                float v = acc[s][e];
                if constexpr (BIAS) v += bias[col];
                if constexpr (RMODE == 1 || RMODE == 2) {
                    v += resid[row_off<(RMODE == 2) ? 1 : 0>(mrow, ldr) + col];
                } else if constexpr (RMODE == 3) {
                    const int t = (mrow / NPATCH) % TT;
                    if (t != 0)
                        v += resid[(size_t)mrow * ldr + col]
                           - resid[(size_t)(mrow - NPATCH) * ldr + col];
                }
                Out[row_off<OMODE>(mrow, ldo) + col] = v;
            }
        }
    }
}

// ------------------------------------------------ spatial: dw3x3 + LN + gelu
__global__ __launch_bounds__(256)
void spatial_dw_norm(const float* __restrict__ ph, const float* __restrict__ wdw,
                     const float* __restrict__ gam, const float* __restrict__ bet,
                     float* __restrict__ out)
{
    const int bt  = blockIdx.x;                 // B*T samples
    const int tid = threadIdx.x;
    const size_t base = (size_t)bt * NPATCH * ADIM;
    float s = 0.f, ss = 0.f;
    for (int idx = tid; idx < NPATCH * ADIM; idx += 256) {
        const int n = idx / ADIM, c = idx - n * ADIM;
        const int h = n / HWD,   w = n - h * HWD;
        float acc = 0.f;
#pragma unroll
        for (int dy = -1; dy <= 1; ++dy)
#pragma unroll
            for (int dx = -1; dx <= 1; ++dx) {
                const int hh = h + dy, ww = w + dx;
                if (hh >= 0 && hh < HWD && ww >= 0 && ww < HWD)
                    acc += ph[base + (size_t)(hh * HWD + ww) * ADIM + c]
                         * wdw[c * 9 + (dy + 1) * 3 + (dx + 1)];
            }
        out[base + idx] = acc;
        s += acc; ss += acc * acc;
    }
    __shared__ float shs[256], shq[256];
    shs[tid] = s; shq[tid] = ss; __syncthreads();
    for (int o = 128; o > 0; o >>= 1) {
        if (tid < o) { shs[tid] += shs[tid + o]; shq[tid] += shq[tid + o]; }
        __syncthreads();
    }
    __shared__ float mu, rstd;
    if (tid == 0) {
        const float inv = 1.0f / (NPATCH * ADIM);
        const float m = shs[0] * inv;
        const float v = shq[0] * inv - m * m;
        mu = m; rstd = rsqrtf(v + 1e-5f);
    }
    __syncthreads();
    for (int idx = tid; idx < NPATCH * ADIM; idx += 256) {
        const int c = idx % ADIM;
        const float v = (out[base + idx] - mu) * rstd * gam[c] + bet[c];
        out[base + idx] = gelu_f(v);
    }
}

// ------------------------------------------- temporal: dw3(T) + norm + gelu
__global__ __launch_bounds__(256)
void temporal_dw_norm(const float* __restrict__ tin, const float* __restrict__ wdw,
                      const float* __restrict__ gam, const float* __restrict__ bet,
                      float* __restrict__ out)
{
    const int bn  = blockIdx.x;                 // B*N samples
    const int b   = bn / NPATCH, n = bn - b * NPATCH;
    const int tid = threadIdx.x;
    float vals[6]; float s = 0.f, ss = 0.f;
#pragma unroll
    for (int j = 0; j < 6; ++j) {               // 6*256 = 1536 = ADIM*TT
        const int idx = tid + j * 256;
        const int c = idx / TT, t = idx - c * TT;
        float acc = 0.f;
#pragma unroll
        for (int k = 0; k < 3; ++k) {
            const int t2 = t + k - 1;
            if (t2 >= 0 && t2 < TT)
                acc += tin[((size_t)(b * TT + t2) * NPATCH + n) * ADIM + c] * wdw[c * 3 + k];
        }
        vals[j] = acc; s += acc; ss += acc * acc;
    }
    __shared__ float shs[256], shq[256];
    shs[tid] = s; shq[tid] = ss; __syncthreads();
    for (int o = 128; o > 0; o >>= 1) {
        if (tid < o) { shs[tid] += shs[tid + o]; shq[tid] += shq[tid + o]; }
        __syncthreads();
    }
    __shared__ float mu, rstd;
    if (tid == 0) {
        const float inv = 1.0f / (ADIM * TT);
        const float m = shs[0] * inv;
        const float v = shq[0] * inv - m * m;
        mu = m; rstd = rsqrtf(v + 1e-5f);
    }
    __syncthreads();
#pragma unroll
    for (int j = 0; j < 6; ++j) {
        const int idx = tid + j * 256;
        const int c = idx / TT, t = idx - c * TT;
        const float v = (vals[j] - mu) * rstd * gam[c] + bet[c];
        out[((size_t)(b * TT + t) * NPATCH + n) * ADIM + c] = gelu_f(v);
    }
}

// ------------------------------------------------------- fused elementwise
__global__ void zero_k(float* __restrict__ p) {
    p[(size_t)blockIdx.x * 256 + threadIdx.x] = 0.f;
}

// tp_in = ph + diff2 (in place on diff2) + accumulate gi = sum |diff2|
__global__ __launch_bounds__(192)
void add_gi_k(const float* __restrict__ ph, float* __restrict__ diff2_io,
              float* __restrict__ gi)
{
    const int bt = blockIdx.x, b = bt / TT, c = threadIdx.x;
    const size_t base = (size_t)bt * NPATCH * ADIM;
    float s = 0.f;
    for (int n = 0; n < NPATCH; ++n) {
        const size_t o = base + (size_t)n * ADIM + c;
        const float v = diff2_io[o];
        s += fabsf(v);
        diff2_io[o] = ph[o] + v;
    }
    atomicAdd(&gi[b * ADIM + c], s);
}

__global__ __launch_bounds__(192)
void gate_mlp_k(const float* __restrict__ gi,
                const float* __restrict__ w1, const float* __restrict__ b1,
                const float* __restrict__ w2, const float* __restrict__ b2,
                float* __restrict__ gate)
{
    const int b = blockIdx.x, c = threadIdx.x;
    __shared__ float sh[ADIM], sh2[ADIM];
    sh[c] = gi[b * ADIM + c] * (1.0f / (TT * NPATCH));
    __syncthreads();
    float a = b1[c];
    for (int k = 0; k < ADIM; ++k) a += sh[k] * w1[k * ADIM + c];
    sh2[c] = gelu_f(a);
    __syncthreads();
    float g = b2[c];
    for (int k = 0; k < ADIM; ++k) g += sh2[k] * w2[k * ADIM + c];
    gate[b * ADIM + c] = 1.0f / (1.0f + expf(-g));
}

// fused = gate*temporal + (1-gate)*spatial, plus clsb[bt][c] = mean_n fused
__global__ __launch_bounds__(192)
void fuse_cls_k(const float* __restrict__ temporal, const float* __restrict__ spatial,
                const float* __restrict__ gate, float* __restrict__ fused,
                float* __restrict__ clsb)
{
    const int bt = blockIdx.x, b = bt / TT, c = threadIdx.x;
    const size_t base = (size_t)bt * NPATCH * ADIM;
    const float g = gate[b * ADIM + c];
    float s = 0.f;
    for (int n = 0; n < NPATCH; ++n) {
        const size_t o = base + (size_t)n * ADIM + c;
        const float f = g * temporal[o] + (1.0f - g) * spatial[o];
        fused[o] = f;
        s += f;
    }
    clsb[bt * ADIM + c] = s * (1.0f / NPATCH);
}

__global__ __launch_bounds__(256)
void cls_out_k(const float* __restrict__ x, const float* __restrict__ clsb,
               const float* __restrict__ wc, const float* __restrict__ bc,
               float* __restrict__ out)
{
    const int bt = blockIdx.x, tid = threadIdx.x;
    __shared__ float sh[ADIM];
    if (tid < ADIM) sh[tid] = clsb[bt * ADIM + tid];
    __syncthreads();
    const size_t off = (size_t)bt * 197 * CDIM;       // CLS row (n = 0)
    for (int o = tid; o < CDIM; o += 256) {
        float a = bc[o];
        for (int k = 0; k < ADIM; ++k) a += sh[k] * wc[(size_t)k * CDIM + o];
        out[off + o] = x[off + o] + a;
    }
}

// ------------------------------------------------------------------- launch
extern "C" void kernel_launch(void* const* d_in, const int* in_sizes, int n_in,
                              void* d_out, int out_size, void* d_ws, size_t ws_size,
                              hipStream_t stream) {
    (void)in_sizes; (void)n_in; (void)out_size; (void)ws_size;

    const float* x      = (const float*)d_in[0];
    const float* w_down = (const float*)d_in[1];
    const float* b_down = (const float*)d_in[2];
    const float* w_sdw  = (const float*)d_in[3];
    const float* w_spw  = (const float*)d_in[4];
    const float* g_sn   = (const float*)d_in[5];
    const float* b_sn   = (const float*)d_in[6];
    const float* w_diff = (const float*)d_in[7];
    const float* w_tdw  = (const float*)d_in[8];
    const float* w_tpw  = (const float*)d_in[9];
    const float* g_tn   = (const float*)d_in[10];
    const float* b_tn   = (const float*)d_in[11];
    const float* w_g1   = (const float*)d_in[12];
    const float* b_g1   = (const float*)d_in[13];
    const float* w_g2   = (const float*)d_in[14];
    const float* b_g2   = (const float*)d_in[15];
    const float* w_up   = (const float*)d_in[16];
    const float* b_up   = (const float*)d_in[17];
    const float* w_cls  = (const float*)d_in[18];
    const float* b_cls  = (const float*)d_in[19];
    float* out = (float*)d_out;

    float* ph   = (float*)d_ws;
    float* bufA = ph   + MA;   // activations (spatial act / temporal act / fused)
    float* bufS = bufA + MA;   // spatial branch output
    float* bufT = bufS + MA;   // diff2 -> tp_in -> temporal branch output
    float* gi   = bufT + MA;
    float* gate = gi   + BB * ADIM;
    float* clsb = gate + BB * ADIM;
    // packed bf16 weights (fragment order), after the float region
    __bf16* p_down = (__bf16*)(clsb + (size_t)BB * TT * ADIM);
    __bf16* p_spw  = p_down + (size_t)CDIM * ADIM;   // 768x192
    __bf16* p_diff = p_spw  + (size_t)ADIM * ADIM;
    __bf16* p_tpw  = p_diff + (size_t)ADIM * ADIM;
    __bf16* p_up   = p_tpw  + (size_t)ADIM * ADIM;   // 192x768

    // 0) pre-pack all GEMM weights into WMMA fragment order (tiny, L2-resident)
    pack_b_k<0><<<(CDIM * ADIM) / 256, 256, 0, stream>>>(w_down, ADIM, p_down, CDIM, CDIM * ADIM);
    pack_b_k<1><<<(ADIM * ADIM) / 256, 256, 0, stream>>>(w_spw,  ADIM, p_spw,  ADIM, ADIM * ADIM);
    pack_b_k<0><<<(ADIM * ADIM) / 256, 256, 0, stream>>>(w_diff, ADIM, p_diff, ADIM, ADIM * ADIM);
    pack_b_k<1><<<(ADIM * ADIM) / 256, 256, 0, stream>>>(w_tpw,  ADIM, p_tpw,  ADIM, ADIM * ADIM);
    pack_b_k<0><<<(ADIM * CDIM) / 256, 256, 0, stream>>>(w_up,   CDIM, p_up,   ADIM, ADIM * CDIM);

    const dim3 gP(MROWS / 32);       // 1568 row-panel blocks

    // 1) ph = patch @ w_down + b_down        (x read exactly once)
    gemm_panel_wmma<CDIM, 1, 1, 0, 0><<<gP, 256, 0, stream>>>(
        x, CDIM, p_down, b_down, nullptr, 0, ph, ADIM, ADIM);

    // 2) spatial depthwise 3x3 + per-sample norm + gelu -> bufA
    spatial_dw_norm<<<BB * TT, 256, 0, stream>>>(ph, w_sdw, g_sn, b_sn, bufA);

    // 3) spatial = act @ w_spw^T -> bufS
    gemm_panel_wmma<ADIM, 0, 0, 0, 0><<<gP, 256, 0, stream>>>(
        bufA, ADIM, p_spw, nullptr, nullptr, 0, bufS, ADIM, ADIM);

    // 4) diff2 = diff + diff @ w_diff -> bufT   (diff formed on the fly from ph)
    gemm_panel_wmma<ADIM, 2, 0, 3, 0><<<gP, 256, 0, stream>>>(
        ph, ADIM, p_diff, nullptr, ph, ADIM, bufT, ADIM, ADIM);

    // 5) gi = sum |diff2| (fused with tp_in = ph + diff2), then gate MLP
    zero_k<<<(BB * ADIM) / 256, 256, 0, stream>>>(gi);
    add_gi_k<<<BB * TT, ADIM, 0, stream>>>(ph, bufT, gi);
    gate_mlp_k<<<BB, ADIM, 0, stream>>>(gi, w_g1, b_g1, w_g2, b_g2, gate);

    // 6) temporal depthwise(T) + per-(b,n) norm + gelu -> bufA
    temporal_dw_norm<<<BB * NPATCH, 256, 0, stream>>>(bufT, w_tdw, g_tn, b_tn, bufA);

    // 7) temporal = act @ w_tpw^T -> bufT
    gemm_panel_wmma<ADIM, 0, 0, 0, 0><<<gP, 256, 0, stream>>>(
        bufA, ADIM, p_tpw, nullptr, nullptr, 0, bufT, ADIM, ADIM);

    // 8) fused = gate*temporal + (1-gate)*spatial -> bufA, + CLS means
    fuse_cls_k<<<BB * TT, ADIM, 0, stream>>>(bufT, bufS, gate, bufA, clsb);

    // 9) patch_out = patch + fused @ w_up + b_up  (fused read exactly once)
    gemm_panel_wmma<ADIM, 0, 1, 2, 1><<<gP, 256, 0, stream>>>(
        bufA, ADIM, p_up, b_up, x, CDIM, out, CDIM, CDIM);

    // 10) cls_out = cls_token + cls_ctx @ w_cls + b_cls
    cls_out_k<<<BB * TT, 256, 0, stream>>>(x, clsb, w_cls, b_cls, out);
}
